// AttnBlock_48292612276812
// MI455X (gfx1250) — matrix operand order repaired and checked by reading, compiled, not run on hardware
//
#include <hip/hip_runtime.h>
#include <hip/hip_bf16.h>

// ---------------------------------------------------------------------------
// Self-attention block for MI455X (gfx1250), wave32 + WMMA f16->f32.
//   qkv = X @ Wqkv^T + b ; per-head attention ; out = A @ Wout^T + b
// GEMMs: double-buffered LDS tiles filled with GLOBAL_LOAD_ASYNC_TO_LDS_B128
// (ASYNCcnt) when the builtins are available; v_wmma_f32_16x16x32_f16 compute.
// ---------------------------------------------------------------------------

typedef __attribute__((ext_vector_type(16))) _Float16 v16h;
typedef __attribute__((ext_vector_type(8)))  _Float16 v8h;
typedef __attribute__((ext_vector_type(8)))  float    v8f;
typedef __attribute__((ext_vector_type(4)))  int      i32x4;

#define NB   16
#define NS   1024
#define NC   512
#define NH   8
#define ND   64
#define NM   (NB * NS)      // 16384 token rows
#define NQKV (3 * NC)       // 1536

// ---- CDNA5 async global->LDS path (guarded; falls back to ds_store copy) --
#if defined(__has_builtin)
#if __has_builtin(__builtin_amdgcn_global_load_async_to_lds_b128) && \
    __has_builtin(__builtin_amdgcn_s_wait_asynccnt)
#define USE_ASYNC_LDS 1
#endif
#endif
#ifndef USE_ASYNC_LDS
#define USE_ASYNC_LDS 0
#endif

// Builtin takes typed pointers: (int4 as1* gsrc, int4 as3* ldst, imm, imm).
typedef __attribute__((address_space(1))) i32x4 gi32x4;
typedef __attribute__((address_space(3))) i32x4 li32x4;

static __device__ __forceinline__ void copy16_g2l(const _Float16* g, _Float16* l) {
#if USE_ASYNC_LDS
  __builtin_amdgcn_global_load_async_to_lds_b128(
      (gi32x4*)(uintptr_t)g,                 // 64-bit global address
      (li32x4*)(uint32_t)(uintptr_t)l,       // low 32 bits = LDS byte offset
      0, 0);
#else
  *(v8h*)l = *(const v8h*)g;
#endif
}

static __device__ __forceinline__ void wait_stage() {
#if USE_ASYNC_LDS
  __builtin_amdgcn_s_wait_asynccnt(0);   // s_wait_asynccnt 0
#endif
}

static __device__ __forceinline__ v16h pack16(v8h lo, v8h hi) {
  v16h r;
#pragma unroll
  for (int i = 0; i < 8; ++i) { r[i] = lo[i]; r[i + 8] = hi[i]; }
  return r;
}

static __device__ __forceinline__ v8f wmma_f16(v16h a, v16h b, v8f c) {
  // (neg_a, A, neg_b, B, c_mod, C, reuse_a, reuse_b)
  return __builtin_amdgcn_wmma_f32_16x16x32_f16(false, a, false, b, (short)0, c,
                                                false, false);
}

// ------------------------------- fp32 -> fp16 ------------------------------
__global__ void cvt_f32_to_f16(const float* __restrict__ s,
                               _Float16* __restrict__ d, int n) {
  int i = blockIdx.x * blockDim.x + threadIdx.x;
  int stride = gridDim.x * blockDim.x;
  for (; i < n; i += stride) d[i] = (_Float16)s[i];
}

// ------------------------ Kernel 1: fused QKV GEMM -------------------------
// X16 [16384,512] f16 (k-contig), W1 [1536,512] f16 (k-contig).
// Tile 128x128x64, 256 threads = 8 waves (2x4 wave grid), each wave 64x32.
// Double-buffered LDS; async fill of next K-tile overlaps WMMA on current.
// Epilogue: +bias, q*=1/8, scatter to qh/kh [bh][s][d] and vT [bh][d][s].
__global__ __launch_bounds__(256) void qkv_gemm(
    const _Float16* __restrict__ X, const _Float16* __restrict__ W,
    const float* __restrict__ bias, _Float16* __restrict__ qh,
    _Float16* __restrict__ kh, _Float16* __restrict__ vT) {
  __shared__ __align__(64) _Float16 sA[2][128 * 64];   // 2 x 16 KB
  __shared__ __align__(64) _Float16 sB[2][128 * 64];   // 2 x 16 KB

  const int bm = blockIdx.x * 128;
  const int bn = blockIdx.y * 128;
  const int t = threadIdx.x;
  const int lane = t & 31, wid = t >> 5;
  const int waveM = wid >> 2, waveN = wid & 3;
  const int lm = lane & 15, lhi = lane >> 4;

  v8f acc[4][2];
#pragma unroll
  for (int i = 0; i < 4; ++i)
#pragma unroll
    for (int j = 0; j < 2; ++j) acc[i][j] = (v8f){};

  const int ldRow = t >> 1;            // 0..127
  const int ldCol = (t & 1) * 32;      // 0 or 32 elems
  const _Float16* gA = X + (bm + ldRow) * NC + ldCol;
  const _Float16* gB = W + (bn + ldRow) * NC + ldCol;
  const int lOff = ldRow * 64 + ldCol;

  // Prologue: stage K-tile 0 into buffer 0.
#pragma unroll
  for (int u = 0; u < 4; ++u) {
    copy16_g2l(gA + u * 8, &sA[0][lOff + u * 8]);
    copy16_g2l(gB + u * 8, &sB[0][lOff + u * 8]);
  }
  wait_stage();
  __syncthreads();

  for (int kt = 0; kt < 8; ++kt) {
    const int cur = kt & 1;
    if (kt < 7) {                      // async fill of the other buffer
      const _Float16* pa = gA + (kt + 1) * 64;
      const _Float16* pb = gB + (kt + 1) * 64;
#pragma unroll
      for (int u = 0; u < 4; ++u) {
        copy16_g2l(pa + u * 8, &sA[1 - cur][lOff + u * 8]);
        copy16_g2l(pb + u * 8, &sB[1 - cur][lOff + u * 8]);
      }
#if !USE_ASYNC_LDS
      __builtin_prefetch(pa, 0, 3);
      __builtin_prefetch(pb, 0, 3);
#endif
    }
    const _Float16* aBase = &sA[cur][(waveM * 64 + lm) * 64 + lhi * 16];
    const _Float16* bBase = &sB[cur][(waveN * 32 + lm) * 64 + lhi * 16];
#pragma unroll
    for (int ks = 0; ks < 2; ++ks) {
      v16h aF[4], bF[2];
#pragma unroll
      for (int i = 0; i < 4; ++i) {
        const _Float16* p = aBase + i * 1024 + ks * 32;
        aF[i] = pack16(*(const v8h*)p, *(const v8h*)(p + 8));
      }
#pragma unroll
      for (int j = 0; j < 2; ++j)
        bF[j] = *(const v16h*)(bBase + j * 1024 + ks * 32);
#pragma unroll
      for (int i = 0; i < 4; ++i)
#pragma unroll
        for (int j = 0; j < 2; ++j) acc[i][j] = wmma_f16(aF[i], bF[j], acc[i][j]);
    }
    if (kt < 7) {
      wait_stage();
      __syncthreads();
    }
  }

  // Epilogue: bias + scatter into head layouts.
#pragma unroll
  for (int i = 0; i < 4; ++i) {
    const int rowBase = bm + waveM * 64 + i * 16 + 8 * lhi;
#pragma unroll
    for (int j = 0; j < 2; ++j) {
      const int col = bn + waveN * 32 + j * 16 + lm;   // [0,1536)
      const float bv = bias[col];
      const int seg = col >> 9;                        // 0=q 1=k 2=v (uniform)
      const int c = col & (NC - 1);
      const int h = c >> 6, dd = c & 63;
#pragma unroll
      for (int r = 0; r < 8; ++r) {
        const int row = rowBase + r;
        const int b = row >> 10, s = row & (NS - 1);
        const int bh = b * NH + h;
        const float v = acc[i][j][r] + bv;
        if (seg == 0)
          qh[(bh * NS + s) * ND + dd] = (_Float16)(v * 0.125f);   // 1/sqrt(64)
        else if (seg == 1)
          kh[(bh * NS + s) * ND + dd] = (_Float16)v;
        else
          vT[(bh * ND + dd) * NS + s] = (_Float16)v;              // transposed
      }
    }
  }
}

// -------------------- Kernel 2: attention (scores+softmax+PV) --------------
// One block per (b,h,16-row tile). 128 threads = 4 waves.
// Full 16x1024 fp32 score strip in LDS (64KB) -> softmax -> WMMA with V^T.
__global__ __launch_bounds__(128) void attn_kernel(
    const _Float16* __restrict__ qh, const _Float16* __restrict__ kh,
    const _Float16* __restrict__ vT, _Float16* __restrict__ ao) {
  __shared__ __align__(64) float sc[16 * NS];   // 64 KB of 320 KB/WGP

  const int t = threadIdx.x, lane = t & 31, wid = t >> 5;
  const int lm = lane & 15, lhi = lane >> 4;
  const int mBlk = blockIdx.x & 63;          // S/16 tiles
  const int bh = blockIdx.x >> 6;
  const int b = bh >> 3, h = bh & 7;
  const int sBase = mBlk * 16;

  // Q fragments for the 16-row tile (K = D = 64 -> 2 k-steps), from HBM/L2.
  v16h qF[2];
  {
    const _Float16* qp = qh + (bh * NS + sBase + lm) * ND + lhi * 16;
#pragma unroll
    for (int ks = 0; ks < 2; ++ks)
      qF[ks] = pack16(*(const v8h*)(qp + ks * 32), *(const v8h*)(qp + ks * 32 + 8));
  }

  // Phase A: S = Q @ K^T, each wave owns 16 N-tiles (256 columns).
  const _Float16* kbase = kh + (size_t)bh * NS * ND;
  for (int nt = wid * 16; nt < wid * 16 + 16; ++nt) {
    v8f acc = (v8f){};
#pragma unroll
    for (int ks = 0; ks < 2; ++ks) {
      const _Float16* bp = kbase + (nt * 16 + lm) * ND + ks * 32 + lhi * 16;
      acc = wmma_f16(qF[ks], *(const v16h*)bp, acc);
    }
#pragma unroll
    for (int r = 0; r < 8; ++r)
      sc[(r + 8 * lhi) * NS + nt * 16 + lm] = acc[r];
  }
  __syncthreads();

  // Phase B: softmax over 1024 cols; 8 lanes per row, shuffle reductions.
  {
    const int row = t >> 3;
    float* p = sc + row * NS + (t & 7) * 128;
    float mx = -3.0e38f;
    for (int i = 0; i < 128; ++i) mx = fmaxf(mx, p[i]);
    mx = fmaxf(mx, __shfl_xor(mx, 1, 32));
    mx = fmaxf(mx, __shfl_xor(mx, 2, 32));
    mx = fmaxf(mx, __shfl_xor(mx, 4, 32));
    float sum = 0.f;
    for (int i = 0; i < 128; ++i) { float e = __expf(p[i] - mx); p[i] = e; sum += e; }
    sum += __shfl_xor(sum, 1, 32);
    sum += __shfl_xor(sum, 2, 32);
    sum += __shfl_xor(sum, 4, 32);
    const float inv = 1.0f / sum;
    for (int i = 0; i < 128; ++i) p[i] *= inv;
  }
  __syncthreads();

  // Phase C: O = P @ V. Wave wid owns d-tile wid; V^T gives contiguous-k B frags.
  const int nd = wid;
  const _Float16* vbase = vT + ((size_t)bh * ND + nd * 16 + lm) * NS + lhi * 16;
  v8f o = (v8f){};
  for (int ks = 0; ks < 32; ++ks) {
    const float* ap = sc + lm * NS + ks * 32 + lhi * 16;
    v8f p0 = *(const v8f*)ap;
    v8f p1 = *(const v8f*)(ap + 8);
    v16h af;
#pragma unroll
    for (int q = 0; q < 8; ++q) { af[q] = (_Float16)p0[q]; af[q + 8] = (_Float16)p1[q]; }
    o = wmma_f16(af, *(const v16h*)(vbase + ks * 32), o);
  }
#pragma unroll
  for (int r = 0; r < 8; ++r) {
    const int s = sBase + r + 8 * lhi;
    ao[((size_t)b * NS + s) * NC + h * ND + nd * 16 + lm] = (_Float16)o[r];
  }
}

// ----------------------- Kernel 3: output projection -----------------------
__global__ __launch_bounds__(256) void out_gemm(
    const _Float16* __restrict__ A, const _Float16* __restrict__ W,
    const float* __restrict__ bias, float* __restrict__ out) {
  __shared__ __align__(64) _Float16 sA[2][128 * 64];
  __shared__ __align__(64) _Float16 sB[2][128 * 64];

  const int bm = blockIdx.x * 128;
  const int bn = blockIdx.y * 128;
  const int t = threadIdx.x;
  const int lane = t & 31, wid = t >> 5;
  const int waveM = wid >> 2, waveN = wid & 3;
  const int lm = lane & 15, lhi = lane >> 4;

  v8f acc[4][2];
#pragma unroll
  for (int i = 0; i < 4; ++i)
#pragma unroll
    for (int j = 0; j < 2; ++j) acc[i][j] = (v8f){};

  const int ldRow = t >> 1;
  const int ldCol = (t & 1) * 32;
  const _Float16* gA = A + (bm + ldRow) * NC + ldCol;
  const _Float16* gB = W + (bn + ldRow) * NC + ldCol;
  const int lOff = ldRow * 64 + ldCol;

#pragma unroll
  for (int u = 0; u < 4; ++u) {
    copy16_g2l(gA + u * 8, &sA[0][lOff + u * 8]);
    copy16_g2l(gB + u * 8, &sB[0][lOff + u * 8]);
  }
  wait_stage();
  __syncthreads();

  for (int kt = 0; kt < 8; ++kt) {
    const int cur = kt & 1;
    if (kt < 7) {
      const _Float16* pa = gA + (kt + 1) * 64;
      const _Float16* pb = gB + (kt + 1) * 64;
#pragma unroll
      for (int u = 0; u < 4; ++u) {
        copy16_g2l(pa + u * 8, &sA[1 - cur][lOff + u * 8]);
        copy16_g2l(pb + u * 8, &sB[1 - cur][lOff + u * 8]);
      }
#if !USE_ASYNC_LDS
      __builtin_prefetch(pa, 0, 3);
      __builtin_prefetch(pb, 0, 3);
#endif
    }
    const _Float16* aBase = &sA[cur][(waveM * 64 + lm) * 64 + lhi * 16];
    const _Float16* bBase = &sB[cur][(waveN * 32 + lm) * 64 + lhi * 16];
#pragma unroll
    for (int ks = 0; ks < 2; ++ks) {
      v16h aF[4], bF[2];
#pragma unroll
      for (int i = 0; i < 4; ++i) {
        const _Float16* p = aBase + i * 1024 + ks * 32;
        aF[i] = pack16(*(const v8h*)p, *(const v8h*)(p + 8));
      }
#pragma unroll
      for (int j = 0; j < 2; ++j)
        bF[j] = *(const v16h*)(bBase + j * 1024 + ks * 32);
#pragma unroll
      for (int i = 0; i < 4; ++i)
#pragma unroll
        for (int j = 0; j < 2; ++j) acc[i][j] = wmma_f16(aF[i], bF[j], acc[i][j]);
    }
    if (kt < 7) {
      wait_stage();
      __syncthreads();
    }
  }

#pragma unroll
  for (int i = 0; i < 4; ++i) {
    const int rowBase = bm + waveM * 64 + i * 16 + 8 * lhi;
#pragma unroll
    for (int j = 0; j < 2; ++j) {
      const int col = bn + waveN * 32 + j * 16 + lm;
      const float bv = bias[col];
#pragma unroll
      for (int r = 0; r < 8; ++r)
        out[(size_t)(rowBase + r) * NC + col] = acc[i][j][r] + bv;
    }
  }
}

// ---------------------------------------------------------------------------
extern "C" void kernel_launch(void* const* d_in, const int* in_sizes, int n_in,
                              void* d_out, int out_size, void* d_ws,
                              size_t ws_size, hipStream_t stream) {
  (void)in_sizes; (void)n_in; (void)out_size; (void)ws_size;
  const float* vis   = (const float*)d_in[0];
  // d_in[1] (text_feat) unused: self-attention path.
  const float* w_qkv = (const float*)d_in[2];
  const float* b_qkv = (const float*)d_in[3];
  const float* w_out = (const float*)d_in[4];
  const float* b_out = (const float*)d_in[5];
  float* out = (float*)d_out;

  const size_t nX = (size_t)NM * NC;        // 8388608
  const size_t nW1 = (size_t)NQKV * NC;     // 786432
  const size_t nW2 = (size_t)NC * NC;       // 262144
  _Float16* ws  = (_Float16*)d_ws;
  _Float16* X16 = ws;
  _Float16* W1  = X16 + nX;
  _Float16* W2  = W1 + nW1;
  _Float16* QH  = W2 + nW2;                 // [B*H][S][D]
  _Float16* KH  = QH + nX;                  // [B*H][S][D]
  _Float16* VT  = KH + nX;                  // [B*H][D][S]
  _Float16* AO  = VT + nX;                  // [B][S][C]

  cvt_f32_to_f16<<<4096, 256, 0, stream>>>(vis, X16, (int)nX);
  cvt_f32_to_f16<<<768, 256, 0, stream>>>(w_qkv, W1, (int)nW1);
  cvt_f32_to_f16<<<256, 256, 0, stream>>>(w_out, W2, (int)nW2);

  dim3 g1(NM / 128, NQKV / 128);            // 128 x 12
  qkv_gemm<<<g1, 256, 0, stream>>>(X16, W1, b_qkv, QH, KH, VT);

  attn_kernel<<<NB * NH * (NS / 16), 128, 0, stream>>>(QH, KH, VT, AO);

  dim3 g3(NM / 128, NC / 128);              // 128 x 4
  out_gemm<<<g3, 256, 0, stream>>>(AO, W2, b_out, out);
}
